// MultiHeadSelfAttention_18519898980788
// MI455X (gfx1250) — matrix operand order repaired and checked by reading, not compiled, not run on hardware
//
#include <hip/hip_runtime.h>
#include <hip/hip_bf16.h>
#include <stdint.h>

// Problem constants (from the reference): B=4, S=2048, D=1024, H=16, d_k=64
#define Bb 4
#define Ss 2048
#define Dd 1024
#define Hh 16
#define DKk 64
#define Mm (Bb * Ss)   // 8192 flattened rows

typedef __bf16 bf16;
typedef __attribute__((ext_vector_type(16))) __bf16 v16bf;
typedef __attribute__((ext_vector_type(8)))  float  v8f;

// ---------------- helpers ----------------

__device__ __forceinline__ bf16 f2bf(float f) {
  union { float f; uint32_t u; } a; a.f = f;
  uint32_t r = a.u + 0x7FFFu + ((a.u >> 16) & 1u);   // round-to-nearest-even
  union { uint16_t u; bf16 b; } o; o.u = (uint16_t)(r >> 16);
  return o.b;
}

__device__ __forceinline__ v8f vzero() {
  v8f z = {0.f, 0.f, 0.f, 0.f, 0.f, 0.f, 0.f, 0.f};
  return z;
}

__device__ __forceinline__ v8f wmma_bf16(v16bf a, v16bf b, v8f c) {
  // (neg_a, A, neg_b, B, c_mod, C, reuse_a, reuse_b)
  return __builtin_amdgcn_wmma_f32_16x16x32_bf16(false, a, false, b, (short)0, c,
                                                 false, false);
}

// A-operand 16x32 tile from row-major [rows, K] storage.
// Lane L (0..15): row r0+L, K = k0+0..15
// Lane L (16..31): row r0+L-16, K = k0+16..31
__device__ __forceinline__ v16bf load_a_tile(const bf16* __restrict__ base, int ld,
                                             int r0, int k0, int lane) {
  const bf16* p = base + (size_t)(r0 + (lane & 15)) * ld + k0 + ((lane >> 4) << 4);
  union { v16bf v; uint4 q[2]; } u;
  u.q[0] = *(const uint4*)(p);
  u.q[1] = *(const uint4*)(p + 8);
  return u.v;
}

// B-operand 32x16 tile for C = A * W^T, W row-major [N, K].
// Lane L (0..15): W row n0+L, K = k0..k0+15 ; lanes 16..31: K = k0+16..k0+31
__device__ __forceinline__ v16bf load_b_tile(const bf16* __restrict__ base, int ld,
                                             int n0, int k0, int lane) {
  const bf16* p = base + (size_t)(n0 + (lane & 15)) * ld + k0 + ((lane >> 4) << 4);
  union { v16bf v; uint4 q[2]; } u;
  u.q[0] = *(const uint4*)(p);
  u.q[1] = *(const uint4*)(p + 8);
  return u.v;
}

// B-operand tile read from an LDS-staged [64][32] bf16 weight block.
__device__ __forceinline__ v16bf load_b_lds(const bf16* base, int n0, int lane) {
  const bf16* p = base + (n0 + (lane & 15)) * 32 + ((lane >> 4) << 4);
  union { v16bf v; uint4 q[2]; } u;
  u.q[0] = *(const uint4*)(p);
  u.q[1] = *(const uint4*)(p + 8);
  return u.v;
}

// Async global->LDS copy of one 64x32 bf16 weight tile (4 KB), block-cooperative:
// 128 threads x 2 transfers x 16 B. Tracked by ASYNCcnt (no VGPR return path).
// Flat shared pointers carry the LDS byte offset in their low 32 bits
__device__ __forceinline__ void stage_w_tile(const bf16* __restrict__ W, int col0,
                                             int k0, bf16* lbuf, int tid) {
#pragma unroll
  for (int i = 0; i < 2; ++i) {
    const int tr = tid * 2 + i;   // 0..255
    const int r  = tr >> 2;       // row within the 64-col strip
    const int q4 = tr & 3;        // 16-byte quarter of the 32-element row
    const bf16* g = W + (size_t)(col0 + r) * Dd + k0 + q4 * 8;
    const uint32_t l = (uint32_t)(uintptr_t)(lbuf + r * 32 + q4 * 8);
    asm volatile("global_load_async_to_lds_b128 %0, %1, off"
                 :: "v"(l), "v"(g) : "memory");
  }
}

__device__ __forceinline__ void wait_async0() {
  asm volatile("s_wait_asynccnt 0" ::: "memory");
}

// ---------------- kernel 0: f32 -> bf16 convert ----------------

__global__ void cvt_kernel(const float* __restrict__ src, bf16* __restrict__ dst,
                           int n4) {
  int i = blockIdx.x * blockDim.x + threadIdx.x;
  if (i < n4) {
    float4 v = ((const float4*)src)[i];
    union { bf16 b[4]; uint2 u; } o;
    o.b[0] = f2bf(v.x); o.b[1] = f2bf(v.y); o.b[2] = f2bf(v.z); o.b[3] = f2bf(v.w);
    ((uint2*)dst)[i] = o.u;
  }
}

// ---------------- kernel 1: QKV projection + fused RoPE ----------------
// grid = (Mm/64, Hh, 3), block = 128 (4 waves). Block tile: 64 rows x 64 cols;
// each wave owns 16 rows. The 64x64 weight strip is identical for all 4 waves,
// so it is staged through LDS with double-buffered async copies.
// z selects Wq/Wk/Wv; z<2 applies RoPE, z==2 writes V^T.

__global__ __launch_bounds__(128) void qkv_rope_kernel(
    const bf16* __restrict__ X, const bf16* __restrict__ Wq,
    const bf16* __restrict__ Wk, const bf16* __restrict__ Wv,
    const int* __restrict__ pos,
    bf16* __restrict__ Q, bf16* __restrict__ K, bf16* __restrict__ Vt) {
  __shared__ __align__(16) bf16 lbuf[2][64 * 32];   // double-buffered W tile, 8 KB

  const int tid  = threadIdx.x;
  const int lane = tid & 31;
  const int wid  = tid >> 5;
  const int row0 = (blockIdx.x * 4 + wid) * 16;
  const int h    = blockIdx.y;
  const int col0 = h * 64;
  const int z    = blockIdx.z;
  const bf16* W  = (z == 0) ? Wq : (z == 1) ? Wk : Wv;

  v8f acc[4];
#pragma unroll
  for (int t = 0; t < 4; ++t) acc[t] = vzero();

  stage_w_tile(W, col0, 0, lbuf[0], tid);          // prologue copy

  for (int k0 = 0; k0 < Dd; k0 += 32) {
    const int cur = (k0 >> 5) & 1;
    wait_async0();                                  // own async ops done
    __syncthreads();                                // everyone's staged data visible
    if (k0 + 32 < Dd) stage_w_tile(W, col0, k0 + 32, lbuf[cur ^ 1], tid);
    if (k0 + 32 < Dd)
      __builtin_prefetch(X + (size_t)(row0 + (lane & 15)) * Dd + k0 + 32, 0, 3);
    const v16bf a = load_a_tile(X, Dd, row0, k0, lane);
#pragma unroll
    for (int t = 0; t < 4; ++t)
      acc[t] = wmma_bf16(a, load_b_lds(lbuf[cur], t * 16, lane), acc[t]);
    __syncthreads();                                // buf[cur] free for reuse
  }

  const int n     = lane & 15;
  const int half8 = (lane >> 4) * 8;

  if (z < 2) {
    bf16* Out = (z == 0) ? Q : K;
    const float kLogTheta = 11.512925464970229f;  // ln(100000)
#pragma unroll
    for (int t = 0; t < 4; ++t) {
      const int f = t * 16 + n;
      const float freq = __expf(-((float)(f & ~1) * (1.0f / 64.f)) * kLogTheta);
#pragma unroll
      for (int j = 0; j < 8; ++j) {
        const int gm   = row0 + half8 + j;
        const int bb   = gm >> 11;        // / Ss
        const int srow = gm & (Ss - 1);
        float sn, cs;
        __sincosf((float)pos[srow] * freq, &sn, &cs);
        const float val = acc[t][j];
        const float oth = __shfl_xor(val, 1);   // partner in the (even,odd) pair
        const float res = (f & 1) ? (oth * sn + val * cs)   // r2 = x1*sin + x2*cos
                                  : (val * cs - oth * sn);  // r1 = x1*cos - x2*sin
        Out[(((size_t)bb * Hh + h) * Ss + srow) * DKk + f] = f2bf(res);
      }
    }
  } else {
    // V stored transposed: Vt[b][h][f][s] so PV becomes an A*B^T GEMM.
#pragma unroll
    for (int t = 0; t < 4; ++t) {
      const int f = t * 16 + n;
#pragma unroll
      for (int j = 0; j < 8; ++j) {
        const int gm   = row0 + half8 + j;
        const int bb   = gm >> 11;
        const int srow = gm & (Ss - 1);
        Vt[(((size_t)bb * Hh + h) * DKk + f) * Ss + srow] = f2bf(acc[t][j]);
      }
    }
  }
}

// ---------------- kernel 2: causal flash attention ----------------
// One wave per (b, h, 16-query tile). Online softmax over 32-key chunks.
// Wave-local only (causal trip counts diverge between waves -> no block
// barriers); P is transposed C-layout -> A-layout through per-wave LDS.
// grid = (B*H*(S/16))/4 = 2048 blocks, block = 128 (4 independent waves).

__global__ __launch_bounds__(128) void attn_kernel(
    const bf16* __restrict__ Q, const bf16* __restrict__ K,
    const bf16* __restrict__ Vt, bf16* __restrict__ O) {
  __shared__ __align__(16) bf16 plds[4][16][32];  // per-wave P transpose buffer

  const int lane = threadIdx.x & 31;
  const int wid  = threadIdx.x >> 5;
  const int gw   = blockIdx.x * 4 + wid;
  const int qt   = gw & (Ss / 16 - 1);
  const int bh   = gw >> 7;
  const int qr0  = qt * 16;

  const bf16* Qp = Q  + (size_t)bh * Ss * DKk;
  const bf16* Kp = K  + (size_t)bh * Ss * DKk;
  const bf16* Vp = Vt + (size_t)bh * DKk * Ss;

  const v16bf qa0 = load_a_tile(Qp, DKk, qr0, 0, lane);
  const v16bf qa1 = load_a_tile(Qp, DKk, qr0, 32, lane);

  v8f acc[4];
#pragma unroll
  for (int t = 0; t < 4; ++t) acc[t] = vzero();
  float mrow[8], lrow[8];
#pragma unroll
  for (int j = 0; j < 8; ++j) { mrow[j] = -1e30f; lrow[j] = 0.f; }

  const int n     = lane & 15;
  const int half8 = (lane >> 4) * 8;
  const int cmax  = (qr0 + 15) >> 5;  // last 32-key chunk touching the diagonal

  for (int c = 0; c <= cmax; ++c) {
    const int kb = c * 32;
    // scores S = Q K^T (16x32 in two 16x16 C tiles), K rows are B-operand rows
    v8f s0 = vzero(), s1 = vzero();
    s0 = wmma_bf16(qa0, load_b_tile(Kp, DKk, kb,       0, lane), s0);
    s0 = wmma_bf16(qa1, load_b_tile(Kp, DKk, kb,      32, lane), s0);
    s1 = wmma_bf16(qa0, load_b_tile(Kp, DKk, kb + 16,  0, lane), s1);
    s1 = wmma_bf16(qa1, load_b_tile(Kp, DKk, kb + 16, 32, lane), s1);

#pragma unroll
    for (int j = 0; j < 8; ++j) {
      const int qi = qr0 + half8 + j;
      float v0 = s0[j] * 0.125f;  if (kb + n      > qi) v0 = -1e30f;  // 1/sqrt(64)
      float v1 = s1[j] * 0.125f;  if (kb + 16 + n > qi) v1 = -1e30f;
      // row max across the 16 lanes of this half-wave
      float mx = fmaxf(v0, v1);
      mx = fmaxf(mx, __shfl_xor(mx, 1));
      mx = fmaxf(mx, __shfl_xor(mx, 2));
      mx = fmaxf(mx, __shfl_xor(mx, 4));
      mx = fmaxf(mx, __shfl_xor(mx, 8));
      const float mnew  = fmaxf(mrow[j], mx);
      const float alpha = __expf(mrow[j] - mnew);
      const float p0    = __expf(v0 - mnew);
      const float p1    = __expf(v1 - mnew);
      float rs = p0 + p1;
      rs += __shfl_xor(rs, 1);
      rs += __shfl_xor(rs, 2);
      rs += __shfl_xor(rs, 4);
      rs += __shfl_xor(rs, 8);
      lrow[j] = lrow[j] * alpha + rs;
      mrow[j] = mnew;
#pragma unroll
      for (int t = 0; t < 4; ++t) acc[t][j] *= alpha;
      // stash P (C-layout element (m = half8+j, key = n / 16+n)) for transpose
      plds[wid][half8 + j][n]      = f2bf(p0);
      plds[wid][half8 + j][16 + n] = f2bf(p1);
    }

    // wave-local LDS ordering: stores above must land before A-layout reads
    asm volatile("s_wait_dscnt 0" ::: "memory");

    v16bf pa;
    {
      const int r    = lane & 15;
      const int koff = (lane >> 4) * 16;
      union { v16bf v; uint4 q[2]; } u;
      u.q[0] = *(const uint4*)&plds[wid][r][koff];
      u.q[1] = *(const uint4*)&plds[wid][r][koff + 8];
      pa = u.v;
    }

    // O += P * V  (V^T rows are features -> contiguous B-operand loads)
#pragma unroll
    for (int t = 0; t < 4; ++t)
      acc[t] = wmma_bf16(pa, load_b_tile(Vp, Ss, t * 16, kb, lane), acc[t]);
  }

  // normalize and write attention output as [B, S, D] bf16
  const int b = bh >> 4, h = bh & 15;
#pragma unroll
  for (int j = 0; j < 8; ++j) {
    const float inv = 1.0f / lrow[j];
    const int srow  = qr0 + half8 + j;
    bf16* o = O + ((size_t)b * Ss + srow) * Dd + h * 64;
#pragma unroll
    for (int t = 0; t < 4; ++t) o[t * 16 + n] = f2bf(acc[t][j] * inv);
  }
}

// ---------------- kernel 3: output projection (f32 result) ----------------
// Same block-cooperative async-LDS weight staging as the QKV kernel.

__global__ __launch_bounds__(128) void out_proj_kernel(
    const bf16* __restrict__ A, const bf16* __restrict__ W,
    float* __restrict__ Y) {
  __shared__ __align__(16) bf16 lbuf[2][64 * 32];

  const int tid  = threadIdx.x;
  const int lane = tid & 31;
  const int wid  = tid >> 5;
  const int row0 = (blockIdx.x * 4 + wid) * 16;
  const int col0 = blockIdx.y * 64;

  v8f acc[4];
#pragma unroll
  for (int t = 0; t < 4; ++t) acc[t] = vzero();

  stage_w_tile(W, col0, 0, lbuf[0], tid);

  for (int k0 = 0; k0 < Dd; k0 += 32) {
    const int cur = (k0 >> 5) & 1;
    wait_async0();
    __syncthreads();
    if (k0 + 32 < Dd) stage_w_tile(W, col0, k0 + 32, lbuf[cur ^ 1], tid);
    if (k0 + 32 < Dd)
      __builtin_prefetch(A + (size_t)(row0 + (lane & 15)) * Dd + k0 + 32, 0, 3);
    const v16bf a = load_a_tile(A, Dd, row0, k0, lane);
#pragma unroll
    for (int t = 0; t < 4; ++t)
      acc[t] = wmma_bf16(a, load_b_lds(lbuf[cur], t * 16, lane), acc[t]);
    __syncthreads();
  }

  const int n     = lane & 15;
  const int half8 = (lane >> 4) * 8;
#pragma unroll
  for (int j = 0; j < 8; ++j) {
    float* y = Y + (size_t)(row0 + half8 + j) * Dd + col0;
#pragma unroll
    for (int t = 0; t < 4; ++t) y[t * 16 + n] = acc[t][j];
  }
}

// ---------------- host launcher ----------------

extern "C" void kernel_launch(void* const* d_in, const int* in_sizes, int n_in,
                              void* d_out, int out_size, void* d_ws, size_t ws_size,
                              hipStream_t stream) {
  const float* X   = (const float*)d_in[0];
  const int*   pos = (const int*)d_in[1];
  const float* Wq  = (const float*)d_in[2];
  const float* Wk  = (const float*)d_in[3];
  const float* Wv  = (const float*)d_in[4];
  const float* Wo  = (const float*)d_in[5];
  float* Y = (float*)d_out;

  char* ws = (char*)d_ws;
  size_t off = 0;
  bf16* Xb  = (bf16*)(ws + off); off += (size_t)Mm * Dd * 2;   // 16 MB
  bf16* Wqb = (bf16*)(ws + off); off += (size_t)Dd * Dd * 2;   //  2 MB
  bf16* Wkb = (bf16*)(ws + off); off += (size_t)Dd * Dd * 2;
  bf16* Wvb = (bf16*)(ws + off); off += (size_t)Dd * Dd * 2;
  bf16* Wob = (bf16*)(ws + off); off += (size_t)Dd * Dd * 2;
  bf16* Qb  = (bf16*)(ws + off); off += (size_t)Mm * Dd * 2;   // [B,H,S,dk]
  bf16* Kb  = (bf16*)(ws + off); off += (size_t)Mm * Dd * 2;   // [B,H,S,dk]
  bf16* Vtb = (bf16*)(ws + off); off += (size_t)Mm * Dd * 2;   // [B,H,dk,S]
  bf16* Ab  = (bf16*)(ws + off); off += (size_t)Mm * Dd * 2;   // attn out [B,S,D]

  // 0) convert everything to bf16
  int n4 = Mm * Dd / 4;
  cvt_kernel<<<(n4 + 255) / 256, 256, 0, stream>>>(X, Xb, n4);
  n4 = Dd * Dd / 4;
  cvt_kernel<<<(n4 + 255) / 256, 256, 0, stream>>>(Wq, Wqb, n4);
  cvt_kernel<<<(n4 + 255) / 256, 256, 0, stream>>>(Wk, Wkb, n4);
  cvt_kernel<<<(n4 + 255) / 256, 256, 0, stream>>>(Wv, Wvb, n4);
  cvt_kernel<<<(n4 + 255) / 256, 256, 0, stream>>>(Wo, Wob, n4);

  // 1) QKV projection + RoPE (z: 0=Q 1=K 2=V)
  qkv_rope_kernel<<<dim3(Mm / 64, Hh, 3), 128, 0, stream>>>(
      Xb, Wqb, Wkb, Wvb, pos, Qb, Kb, Vtb);

  // 2) causal flash attention
  attn_kernel<<<dim3((Bb * Hh * (Ss / 16)) / 4), 128, 0, stream>>>(Qb, Kb, Vtb, Ab);

  // 3) output projection (f32 out)
  out_proj_kernel<<<dim3(Mm / 64, Dd / 64), 128, 0, stream>>>(Ab, Wob, Y);
}
